// CrissCrossAttention_47665547051357
// MI455X (gfx1250) — matrix operand order repaired and checked by reading, not compiled, not run on hardware
//
#include <hip/hip_runtime.h>
#include <hip/hip_bf16.h>
#include <stdint.h>

// ---------------------------------------------------------------------------
// CrissCross attention (height branch), MI455X / gfx1250, wave32 + WMMA bf16.
// Pass 1: Q,K,V = x @ W{q,k,v}^T  -> bf16 workspace (d_ws, 384 MiB).
// Pass 2: per (b,w) column: S = Q K^T (f32 acc), diag-masked softmax,
//         O = P V, out = gamma*O + x.
// ---------------------------------------------------------------------------

typedef __bf16 bf16;
typedef __attribute__((ext_vector_type(16))) bf16  v16bf;
typedef __attribute__((ext_vector_type(8)))  bf16  v8bf;
typedef __attribute__((ext_vector_type(8)))  float v8f;
typedef __attribute__((ext_vector_type(4)))  float f32x4;

#define USE_ASYNC_LDS_COPY 1

#define B_DIM 8
#define H_DIM 128
#define W_DIM 128
#define C_DIM 512
#define M_TOT (B_DIM * H_DIM * W_DIM)   // 131072

// Hardware fp32->bf16 convert (RNE) — let clang pick v_cvt_*bf16* forms.
static __device__ __forceinline__ bf16 f2bf(float f) {
  return static_cast<bf16>(f);
}

static __device__ __forceinline__ v8f vzero8() {
  v8f z;
#pragma unroll
  for (int i = 0; i < 8; ++i) z[i] = 0.0f;
  return z;
}

// Async global->LDS copy (16B per lane). LDS flat addr low 32 bits = LDS offset.
static __device__ __forceinline__ void async_b128(void* lds_dst, const void* gsrc) {
#if USE_ASYNC_LDS_COPY
  unsigned l = (unsigned)(unsigned long long)lds_dst;
  asm volatile("global_load_async_to_lds_b128 %0, %1, off"
               :: "v"(l), "v"(gsrc) : "memory");
#else
  *(v8bf*)lds_dst = *(const v8bf*)gsrc;
#endif
}
static __device__ __forceinline__ void wait_async_le8() {
#if USE_ASYNC_LDS_COPY
  asm volatile("s_wait_asynccnt 0x8" ::: "memory");
#endif
}
static __device__ __forceinline__ void wait_async_0() {
#if USE_ASYNC_LDS_COPY
  asm volatile("s_wait_asynccnt 0x0" ::: "memory");
#endif
}

// A fragment (16x32 bf16). ISA layout: lane<16 -> row=lane, K {0..7, 16..23};
// lane>=16 -> row=lane-16, K {8..15, 24..31}.
static __device__ __forceinline__ v16bf load_frag_a(const bf16* buf, int stride,
                                                    int row_base, int k_base, int lane) {
  int row = row_base + (lane & 15);
  int kh  = (lane >> 4) * 8;
  const bf16* p = buf + row * stride + k_base + kh;
  v8bf lo = *(const v8bf*)p;
  v8bf hi = *(const v8bf*)(p + 16);
  v16bf r;
#pragma unroll
  for (int i = 0; i < 8; ++i) { r[i] = lo[i]; r[8 + i] = hi[i]; }
  return r;
}

// B fragment (32x16 bf16, column n contiguous-over-K in source "row" n).
// lanes 0-15: K {0..7, 16..23} of column n=lane; lanes 16-31: K {8..15, 24..31} of column n=lane-16.
static __device__ __forceinline__ v16bf load_frag_b(const bf16* buf, int stride,
                                                    int row_base, int k_base, int lane) {
  int row = row_base + (lane & 15);
  int kh  = (lane >> 4) * 8;
  const bf16* p = buf + row * stride + k_base + kh;
  v8bf lo = *(const v8bf*)p;
  v8bf hi = *(const v8bf*)(p + 16);
  v16bf r;
#pragma unroll
  for (int i = 0; i < 8; ++i) { r[i] = lo[i]; r[8 + i] = hi[i]; }
  return r;
}

static __device__ __forceinline__ v8f wmma_bf16(v16bf a, v16bf b, v8f c) {
  return __builtin_amdgcn_wmma_f32_16x16x32_bf16(false, a, false, b, (short)0, c,
                                                 false, false);
}

// ---------------------------------------------------------------------------
// Kernel 1: QKV projection GEMM. y[m,n] = sum_c x[m,c] * W[n,c]  (nn.Linear).
// grid (M/128, N/128, 3), block 256 (8 waves). 128x128 tile, K-chunks of 32,
// double-buffered LDS, fp32->bf16 convert on staging.
// ---------------------------------------------------------------------------
__global__ __launch_bounds__(256)
void qkv_gemm_kernel(const float* __restrict__ x,
                     const float* __restrict__ Wq,
                     const float* __restrict__ Wk,
                     const float* __restrict__ Wv,
                     bf16* __restrict__ qkv) {
  extern __shared__ char smem[];
  const int AS = 40;    // staging stride (bf16 elems), bank-conflict padded
  const int OS = 136;   // out staging stride
  // LDS buffers computed (NOT pointer arrays: avoids addrspacecast initializers)
  auto Abuf = [&](int bi) -> bf16* { return (bf16*)(smem +     0 + bi * 10240); };
  auto Bbuf = [&](int bi) -> bf16* { return (bf16*)(smem + 20480 + bi * 10240); };
  bf16* Ob = (bf16*)(smem + 40960);   // total 75776 B

  const int t    = threadIdx.x;
  const int lane = t & 31;
  const int wv   = t >> 5;
  const int wm   = wv & 3;     // 32-row group
  const int wn   = wv >> 2;    // 64-col group
  const int m0   = blockIdx.x * 128;
  const int n0   = blockIdx.y * 128;
  const int sel  = blockIdx.z;
  const float* Wp = (sel == 0) ? Wq : (sel == 1) ? Wk : Wv;
  bf16* out = qkv + (size_t)sel * (size_t)M_TOT * C_DIM;

  const int srow  = t >> 1;    // 0..127
  const int shalf = t & 1;     // 16 floats each

  auto stage = [&](int bi, int kk) {
    const float* xs = x  + (size_t)(m0 + srow) * C_DIM + kk + shalf * 16;
    const float* ws = Wp + (size_t)(n0 + srow) * C_DIM + kk + shalf * 16;
    bf16* ad = Abuf(bi) + srow * AS + shalf * 16;
    bf16* bd = Bbuf(bi) + srow * AS + shalf * 16;
#pragma unroll
    for (int v = 0; v < 2; ++v) {
      f32x4 a0 = *(const f32x4*)(xs + v * 8 + 0);
      f32x4 a1 = *(const f32x4*)(xs + v * 8 + 4);
      f32x4 b0 = *(const f32x4*)(ws + v * 8 + 0);
      f32x4 b1 = *(const f32x4*)(ws + v * 8 + 4);
      v8bf av, bv;
#pragma unroll
      for (int e = 0; e < 4; ++e) {
        av[e] = f2bf(a0[e]); av[4 + e] = f2bf(a1[e]);
        bv[e] = f2bf(b0[e]); bv[4 + e] = f2bf(b1[e]);
      }
      *(v8bf*)(ad + v * 8) = av;
      *(v8bf*)(bd + v * 8) = bv;
    }
  };

  v8f acc[2][4];
#pragma unroll
  for (int i = 0; i < 2; ++i)
#pragma unroll
    for (int j = 0; j < 4; ++j) acc[i][j] = vzero8();

  stage(0, 0);
  __syncthreads();

#pragma unroll 1
  for (int s = 0; s < 16; ++s) {
    const int cur = s & 1;
    if (s + 2 < 16)
      __builtin_prefetch(x + (size_t)(m0 + srow) * C_DIM + (s + 2) * 32, 0, 0);
    if (s + 1 < 16) stage(cur ^ 1, (s + 1) * 32);

    const bf16* Ac = Abuf(cur);
    const bf16* Bc = Bbuf(cur);
    v16bf a[2], b[4];
#pragma unroll
    for (int i = 0; i < 2; ++i)
      a[i] = load_frag_a(Ac, AS, wm * 32 + i * 16, 0, lane);
#pragma unroll
    for (int j = 0; j < 4; ++j)
      b[j] = load_frag_b(Bc, AS, wn * 64 + j * 16, 0, lane);
#pragma unroll
    for (int i = 0; i < 2; ++i)
#pragma unroll
      for (int j = 0; j < 4; ++j)
        acc[i][j] = wmma_bf16(a[i], b[j], acc[i][j]);
    __syncthreads();
  }

  // C/D layout: VGPR r -> M = r + (lane>>4)*8, N = lane&15.
  {
    const int rsel = (lane >> 4) * 8;
    const int col  = lane & 15;
#pragma unroll
    for (int i = 0; i < 2; ++i)
#pragma unroll
      for (int j = 0; j < 4; ++j) {
        const int rb = wm * 32 + i * 16 + rsel;
        const int cb = wn * 64 + j * 16 + col;
#pragma unroll
        for (int r = 0; r < 8; ++r)
          Ob[(rb + r) * OS + cb] = f2bf(acc[i][j][r]);
      }
  }
  __syncthreads();
  {
    bf16* gd = out + (size_t)(m0 + srow) * C_DIM + n0 + shalf * 64;
    const bf16* ls = Ob + srow * OS + shalf * 64;
#pragma unroll
    for (int v = 0; v < 8; ++v)
      *(v8bf*)(gd + v * 8) = *(const v8bf*)(ls + v * 8);
  }
}

// ---------------------------------------------------------------------------
// Kernel 2: per-(b,w) column attention. grid 1024, block 256 (8 waves).
// LDS map (172 KiB): [0..73728) QK double-buffers (phase1) / Vt buffers
// (phase2, reuse); [73728..141312) S (f32, stride 132); [141312..176128) P.
// ---------------------------------------------------------------------------
__global__ __launch_bounds__(256)
void cca_attn_kernel(const bf16* __restrict__ Qm,
                     const bf16* __restrict__ Km,
                     const bf16* __restrict__ Vm,
                     const float* __restrict__ x,
                     const float* __restrict__ gammap,
                     float* __restrict__ out) {
  extern __shared__ char smem[];
  const int QKS = 72, SS = 132, PS = 136, VTS = 136;
  auto Qbuf = [&](int bi) -> bf16* { return (bf16*)(smem +     0 + bi * 18432); };
  auto Kbuf = [&](int bi) -> bf16* { return (bf16*)(smem + 36864 + bi * 18432); };
  auto Vbuf = [&](int bi) -> bf16* { return (bf16*)(smem +     0 + bi * 34816); };
  float* Sb = (float*)(smem + 73728);
  bf16*  Pb = (bf16*)(smem + 141312);

  const int t = threadIdx.x, lane = t & 31, wv = t >> 5;
  const int wm = wv & 3, wn = wv >> 2;
  const int bw = blockIdx.x;
  const int b  = bw >> 7;            // W = 128
  const int w  = bw & 127;
  const size_t colbase = ((size_t)b * H_DIM * W_DIM + w) * C_DIM;  // (b, h=0, w)
  const size_t rstride = (size_t)W_DIM * C_DIM;                    // h -> h+1

  const int srow = t >> 1, shalf = t & 1;

  // ---- staging helpers ----
  auto stage_qk = [&](int bi, int c0) {            // 64-wide C chunk, async
    const bf16* qs = Qm + colbase + (size_t)srow * rstride + c0 + shalf * 32;
    const bf16* ks = Km + colbase + (size_t)srow * rstride + c0 + shalf * 32;
    bf16* qd = Qbuf(bi) + srow * QKS + shalf * 32;
    bf16* kd = Kbuf(bi) + srow * QKS + shalf * 32;
#pragma unroll
    for (int v = 0; v < 4; ++v) {
      async_b128(qd + v * 8, qs + v * 8);
      async_b128(kd + v * 8, ks + v * 8);
    }
  };
  auto stage_v = [&](int bi, int cc) {             // transpose V chunk into LDS
    const bf16* vs = Vm + colbase + (size_t)srow * rstride + cc * 128 + shalf * 64;
    bf16* vt = Vbuf(bi);
    const int g = srow;
#pragma unroll
    for (int v = 0; v < 8; ++v) {
      v8bf d = *(const v8bf*)(vs + v * 8);
      const int c0 = shalf * 64 + v * 8;
#pragma unroll
      for (int e = 0; e < 8; ++e) vt[(c0 + e) * VTS + g] = d[e];
    }
  };

  // ---- Phase 1: S = Q K^T, accumulated in registers ----
  v8f acc[2][4];
#pragma unroll
  for (int i = 0; i < 2; ++i)
#pragma unroll
    for (int j = 0; j < 4; ++j) acc[i][j] = vzero8();

  stage_qk(0, 0);
#pragma unroll 1
  for (int s = 0; s < 8; ++s) {
    const int cur = s & 1;
    if (s + 1 < 8) { stage_qk(cur ^ 1, (s + 1) * 64); wait_async_le8(); }
    else           { wait_async_0(); }
    __syncthreads();
    const bf16* Qc = Qbuf(cur);
    const bf16* Kc = Kbuf(cur);
#pragma unroll
    for (int ks = 0; ks < 64; ks += 32) {
      v16bf a[2], bb[4];
#pragma unroll
      for (int i = 0; i < 2; ++i)
        a[i] = load_frag_a(Qc, QKS, wm * 32 + i * 16, ks, lane);
#pragma unroll
      for (int j = 0; j < 4; ++j)
        bb[j] = load_frag_b(Kc, QKS, wn * 64 + j * 16, ks, lane);
#pragma unroll
      for (int i = 0; i < 2; ++i)
#pragma unroll
        for (int j = 0; j < 4; ++j)
          acc[i][j] = wmma_bf16(a[i], bb[j], acc[i][j]);
    }
    __syncthreads();
  }

  // dump S to LDS
  {
    const int rsel = (lane >> 4) * 8;
    const int col  = lane & 15;
#pragma unroll
    for (int i = 0; i < 2; ++i)
#pragma unroll
      for (int j = 0; j < 4; ++j) {
        const int rb = wm * 32 + i * 16 + rsel;
        const int cb = wn * 64 + j * 16 + col;
#pragma unroll
        for (int r = 0; r < 8; ++r)
          Sb[(rb + r) * SS + cb] = acc[i][j][r];
      }
  }
  __syncthreads();

  // ---- softmax over key-height g with diagonal masked to -inf ----
  const float gamma = *gammap;
  if (t < 128) {
    const int h = t;
    float* sr = Sb + h * SS;
    float mx = -3.4e38f;
#pragma unroll 1
    for (int g = 0; g < 128; ++g)
      if (g != h) mx = fmaxf(mx, sr[g]);
    float sum = 0.0f;
#pragma unroll 1
    for (int g = 0; g < 128; ++g) {
      float e = (g == h) ? 0.0f : __expf(sr[g] - mx);
      sr[g] = e; sum += e;
    }
    const float inv = 1.0f / sum;
    bf16* pr = Pb + h * PS;
#pragma unroll 1
    for (int g = 0; g < 128; ++g) pr[g] = f2bf(sr[g] * inv);
  }
  stage_v(0, 0);                 // region A reuse: phase-1 reads are done
  __syncthreads();

  // ---- Phase 2: O = P V (per 128-wide C chunk), fused epilogue ----
#pragma unroll 1
  for (int cc = 0; cc < 4; ++cc) {
    const int cur = cc & 1;
    if (cc + 1 < 4) stage_v(cur ^ 1, cc + 1);

    v8f oacc[2][4];
#pragma unroll
    for (int i = 0; i < 2; ++i)
#pragma unroll
      for (int j = 0; j < 4; ++j) oacc[i][j] = vzero8();

    const bf16* Vc = Vbuf(cur);
#pragma unroll
    for (int gs = 0; gs < 128; gs += 32) {
      v16bf a[2], bb[4];
#pragma unroll
      for (int i = 0; i < 2; ++i)
        a[i] = load_frag_a(Pb, PS, wm * 32 + i * 16, gs, lane);
#pragma unroll
      for (int j = 0; j < 4; ++j)
        bb[j] = load_frag_b(Vc, VTS, wn * 64 + j * 16, gs, lane);
#pragma unroll
      for (int i = 0; i < 2; ++i)
#pragma unroll
        for (int j = 0; j < 4; ++j)
          oacc[i][j] = wmma_bf16(a[i], bb[j], oacc[i][j]);
    }

    // dump O chunk to Sb (reuse), then coalesced fused store
    {
      const int rsel = (lane >> 4) * 8;
      const int col  = lane & 15;
#pragma unroll
      for (int i = 0; i < 2; ++i)
#pragma unroll
        for (int j = 0; j < 4; ++j) {
          const int rb = wm * 32 + i * 16 + rsel;
          const int cb = wn * 64 + j * 16 + col;
#pragma unroll
          for (int r = 0; r < 8; ++r)
            Sb[(rb + r) * SS + cb] = oacc[i][j][r];
        }
    }
    __syncthreads();
    {
      const int h = srow;
      const size_t gbase = colbase + (size_t)h * rstride + cc * 128 + shalf * 64;
      const float* lo = Sb + h * SS + shalf * 64;
#pragma unroll
      for (int v = 0; v < 16; ++v) {
        f32x4 o  = *(const f32x4*)(lo + v * 4);
        f32x4 xi = *(const f32x4*)(x + gbase + v * 4);
        f32x4 r  = o * gamma + xi;
        *(f32x4*)(out + gbase + v * 4) = r;
      }
    }
    __syncthreads();
  }
}

// ---------------------------------------------------------------------------
extern "C" void kernel_launch(void* const* d_in, const int* in_sizes, int n_in,
                              void* d_out, int out_size, void* d_ws, size_t ws_size,
                              hipStream_t stream) {
  (void)in_sizes; (void)n_in; (void)out_size; (void)ws_size;
  const float* x     = (const float*)d_in[0];
  const float* Wq    = (const float*)d_in[1];
  const float* Wk    = (const float*)d_in[2];
  const float* Wv    = (const float*)d_in[3];
  const float* gamma = (const float*)d_in[4];
  float* out = (float*)d_out;
  bf16*  qkv = (bf16*)d_ws;   // needs 3 * 131072 * 512 * 2 B = 384 MiB

  const size_t lds1 = 75776;
  const size_t lds2 = 176128;
  hipError_t err;
  err = hipFuncSetAttribute((const void*)qkv_gemm_kernel,
                            hipFuncAttributeMaxDynamicSharedMemorySize, (int)lds1);
  (void)err;
  err = hipFuncSetAttribute((const void*)cca_attn_kernel,
                            hipFuncAttributeMaxDynamicSharedMemorySize, (int)lds2);
  (void)err;

  dim3 g1(M_TOT / 128, C_DIM / 128, 3), b1(256);
  qkv_gemm_kernel<<<g1, b1, lds1, stream>>>(x, Wq, Wk, Wv, qkv);

  const bf16* Qp = qkv;
  const bf16* Kp = qkv + (size_t)M_TOT * C_DIM;
  const bf16* Vp = qkv + (size_t)2 * M_TOT * C_DIM;
  dim3 g2(B_DIM * W_DIM), b2(256);
  cca_attn_kernel<<<g2, b2, lds2, stream>>>(Qp, Kp, Vp, x, gamma, out);
}
